// FrostQuantixJAX_87179246174591
// MI455X (gfx1250) — matrix operand order repaired and checked by reading, not compiled, not run on hardware
//
#include <hip/hip_runtime.h>

typedef __attribute__((ext_vector_type(16))) __bf16 v16bf;
typedef __attribute__((ext_vector_type(8)))  __bf16 v8bf;
typedef __attribute__((ext_vector_type(8)))  float  v8f;

#define LAYERS 12
#define BB 8
#define TT 1024
#define EE 768
#define HH 12
#define DD 64
#define VV 50304
#define NTOK (BB * TT) /* 8192 */

// ---------------- bf16 helpers (manual, no dependence on native __bf16 conversions) ---
__device__ __forceinline__ __bf16 f2bf(float f) {
  union { float f; unsigned u; } a; a.f = f;
  unsigned r = a.u + 0x7FFFu + ((a.u >> 16) & 1u);   // round-to-nearest-even
  union { unsigned short s; __bf16 b; } o; o.s = (unsigned short)(r >> 16);
  return o.b;
}

// Build a 16-element bf16 fragment from two contiguous 8-element chunks.
__device__ __forceinline__ v16bf make_frag(const __bf16* p0, const __bf16* p1) {
  v8bf lo = *(const v8bf*)p0;
  v8bf hi = *(const v8bf*)p1;
  v16bf r;
#pragma unroll
  for (int i = 0; i < 8; i++) { r[i] = lo[i]; r[8 + i] = hi[i]; }
  return r;
}

__device__ __forceinline__ float geluf(float x) {
  float x3 = x * x * x;
  return 0.5f * x * (1.f + tanhf(0.7978845608028654f * (x + 0.044715f * x3)));
}
__device__ __forceinline__ float sigmf(float x) { return 1.f / (1.f + __expf(-x)); }

__device__ __forceinline__ float rmax16(float v) {
#pragma unroll
  for (int off = 1; off < 16; off <<= 1) v = fmaxf(v, __shfl_xor(v, off, 32));
  return v;
}
__device__ __forceinline__ float rsum16(float v) {
#pragma unroll
  for (int off = 1; off < 16; off <<= 1) v += __shfl_xor(v, off, 32);
  return v;
}

// ---------------- one-shot weight transpose + f32->bf16: Wt[n*K+k] = bf16(W[k*N+n]) ---
// grid: (N/32, K/32, L); block 256 = (32 lanes x 8 rows, 4 iters)
__global__ __launch_bounds__(256) void wconv_k(const float* __restrict__ W,
                                               __bf16* __restrict__ Wt,
                                               int K, int N) {
  __shared__ float tile[32][33];
  const float* Wl = W + (size_t)blockIdx.z * K * N;
  __bf16* Wtl = Wt + (size_t)blockIdx.z * K * N;
  const int n0 = blockIdx.x * 32, k0 = blockIdx.y * 32;
  const int tx = threadIdx.x & 31, ty = threadIdx.x >> 5;
#pragma unroll
  for (int j = 0; j < 4; j++) {
    int k = ty + j * 8;
    tile[k][tx] = Wl[(size_t)(k0 + k) * N + n0 + tx];   // coalesced read
  }
  __syncthreads();
#pragma unroll
  for (int j = 0; j < 4; j++) {
    int n = ty + j * 8;
    Wtl[(size_t)(n0 + n) * K + k0 + tx] = f2bf(tile[tx][n]);  // coalesced write
  }
}

// ---------------- embedding: x[r,e] = wte[idx[r],e] + wpe[r%T,e] ----------------------
__global__ __launch_bounds__(256) void embed_k(const int* __restrict__ idx,
                                               const float* __restrict__ wte,
                                               const float* __restrict__ wpe,
                                               float* __restrict__ x) {
  size_t i = (size_t)blockIdx.x * 256 + threadIdx.x;
  if (i >= (size_t)NTOK * EE) return;
  int e = (int)(i % EE);
  int tok = (int)(i / EE);
  int t = tok & (TT - 1);
  int id = idx[tok];
  x[i] = wte[(size_t)id * EE + e] + wpe[(size_t)t * EE + e];
}

// ---------------- RMSNorm row kernel -> bf16 ------------------------------------------
__global__ __launch_bounds__(256) void rmsnorm_k(const float* __restrict__ x,
                                                 const float* __restrict__ g,
                                                 __bf16* __restrict__ out) {
  __shared__ float red[256];
  int r = blockIdx.x;
  const float* xr = x + (size_t)r * EE;
  float v[3]; float s = 0.f;
#pragma unroll
  for (int i = 0; i < 3; i++) { v[i] = xr[threadIdx.x + i * 256]; s += v[i] * v[i]; }
  red[threadIdx.x] = s;
  __syncthreads();
  for (int st = 128; st > 0; st >>= 1) {
    if ((int)threadIdx.x < st) red[threadIdx.x] += red[threadIdx.x + st];
    __syncthreads();
  }
  float inv = 1.f / (sqrtf(red[0] * (1.f / EE)) + 1e-8f);
#pragma unroll
  for (int i = 0; i < 3; i++) {
    int e = threadIdx.x + i * 256;
    out[(size_t)r * EE + e] = f2bf(v[i] * inv * g[e]);
  }
}

// ---------------- final LayerNorm on the 8 last-position rows -> bf16 -----------------
__global__ __launch_bounds__(256) void lastln_k(const float* __restrict__ x,
                                                const float* __restrict__ w,
                                                const float* __restrict__ b,
                                                __bf16* __restrict__ out) {
  __shared__ float r1[256], r2[256];
  int bb = blockIdx.x;
  const float* xr = x + ((size_t)bb * TT + (TT - 1)) * EE;
  float v[3]; float s = 0.f, s2 = 0.f;
#pragma unroll
  for (int i = 0; i < 3; i++) {
    v[i] = xr[threadIdx.x + i * 256];
    s += v[i]; s2 += v[i] * v[i];
  }
  r1[threadIdx.x] = s; r2[threadIdx.x] = s2;
  __syncthreads();
  for (int st = 128; st > 0; st >>= 1) {
    if ((int)threadIdx.x < st) { r1[threadIdx.x] += r1[threadIdx.x + st]; r2[threadIdx.x] += r2[threadIdx.x + st]; }
    __syncthreads();
  }
  float mean = r1[0] * (1.f / EE);
  float var  = r2[0] * (1.f / EE) - mean * mean;
  float inv  = rsqrtf(var + 1e-5f);
#pragma unroll
  for (int i = 0; i < 3; i++) {
    int e = threadIdx.x + i * 256;
    out[(size_t)bb * EE + e] = f2bf((v[i] - mean) * inv * w[e] + b[e]);
  }
}

// ---------------- transpose V slice of qkv into vT[b,h,d,t] (bf16) --------------------
__global__ __launch_bounds__(256) void vtrans_k(const __bf16* __restrict__ qkv,
                                                __bf16* __restrict__ vT) {
  size_t i = (size_t)blockIdx.x * 256 + threadIdx.x;
  if (i >= (size_t)NTOK * EE) return;
  int col = (int)(i % EE);       // h*64 + d
  int tok = (int)(i / EE);       // b*T + t
  int b = tok / TT, t = tok & (TT - 1);
  int h = col >> 6, d = col & 63;
  vT[((size_t)(b * HH + h) * DD + d) * TT + t] = qkv[(size_t)tok * (3 * EE) + 2 * EE + col];
}

// ---------------- bf16 WMMA GEMM, no-LDS hot loop, fused epilogues --------------------
// C[M,N] = epi(A_bf16[M,K] x Wt_bf16[N,K]^T + bias)
// Block tile 128(M) x 64(N): 8 waves, wave w owns rows [w*16,+16) x all 64 cols.
// A-fragment reused across 4 WMMAs per 32-K step; B strips served from L0/L2.
enum { EPI_BF16 = 0, EPI_GELU_BF16 = 1, EPI_RESID_F32 = 2, EPI_F32_BF16 = 3, EPI_GATED = 4, EPI_F32 = 5 };

template <int EPI>
__global__ __launch_bounds__(256) void gemm_wmma(const __bf16* __restrict__ A,
                                                 const __bf16* __restrict__ Wt,
                                                 const float* __restrict__ bias,
                                                 float* __restrict__ outF,
                                                 __bf16* __restrict__ outB,
                                                 const float* __restrict__ aux,
                                                 int M, int N, int K) {
  const int tid  = threadIdx.x;
  const int lane = tid & 31;
  const int wave = tid >> 5;
  const int m16  = lane & 15;
  const int sel  = lane >> 4;                     // wave half
  const int bM   = blockIdx.y * 128;
  const int bN   = blockIdx.x * 64;

  int rowA = bM + wave * 16 + m16;
  // A layout (16-bit A): lanes<16 take K 0-7 & 16-23; lanes>=16 take K 8-15 & 24-31.
  const __bf16* ap = A + (size_t)(rowA < M ? rowA : 0) * K + sel * 8;
  // B layout (16-bit B): lane = column; lanes<16 take K 0-7 & 16-23; lanes>=16 take K 8-15 & 24-31.
  const __bf16* bp[4];
#pragma unroll
  for (int t = 0; t < 4; t++)
    bp[t] = Wt + (size_t)(bN + t * 16 + m16) * K + sel * 8;

  v8f acc[4];
#pragma unroll
  for (int t = 0; t < 4; t++)
#pragma unroll
    for (int i = 0; i < 8; i++) acc[t][i] = 0.f;

  for (int kk = 0; kk < K; kk += 32) {
    v16bf af = make_frag(ap, ap + 16);
    __builtin_prefetch(ap + 128, 0, 0);           // -> global_prefetch_b8 (A stream)
#pragma unroll
    for (int t = 0; t < 4; t++) {
      v16bf bf = make_frag(bp[t], bp[t] + 16);
      acc[t] = __builtin_amdgcn_wmma_f32_16x16x32_bf16(false, af, false, bf, (short)0, acc[t], false, false);
      bp[t] += 32;
    }
    ap += 32;
  }

  // Epilogue. C layout: lane n = m16 (+t*16), vgpr i -> row sel*8+i.
#pragma unroll
  for (int t = 0; t < 4; t++) {
    int col = bN + t * 16 + m16;
    float bia = (bias != nullptr) ? bias[col] : 0.f;
#pragma unroll
    for (int i = 0; i < 8; i++) {
      int row = bM + wave * 16 + sel * 8 + i;
      if (row >= M) continue;
      float v = acc[t][i] + bia;
      size_t o = (size_t)row * N + col;
      if (EPI == EPI_BF16)            outB[o] = f2bf(v);
      else if (EPI == EPI_GELU_BF16)  outB[o] = f2bf(geluf(v));
      else if (EPI == EPI_RESID_F32)  outF[o] = outF[o] + v;
      else if (EPI == EPI_F32_BF16) { outF[o] = v; outB[o] = f2bf(v); }
      else if (EPI == EPI_GATED)      outF[o] = outF[o] + aux[o] * sigmf(v);
      else                            outF[o] = v;
    }
  }
}

// ---------------- flash attention: 1 wave per (b, h, 16-query tile) -------------------
__global__ __launch_bounds__(32) void attn_k(const __bf16* __restrict__ qkv,
                                             const __bf16* __restrict__ vT,
                                             __bf16* __restrict__ y) {
  __shared__ __bf16 P[16][40];                    // probs tile, padded
  const int lane = threadIdx.x;
  const int m16  = lane & 15;
  const int sel  = lane >> 4;
  const int qt   = blockIdx.x & 63;               // T/16 = 64 tiles
  const int bh   = blockIdx.x >> 6;
  const int h    = bh % HH;
  const int b    = bh / HH;
  const int q_base = qt * 16;
  const float scale = 0.125f;                     // D^-0.5

  const __bf16* qrow = qkv + (size_t)(b * TT + q_base + m16) * (3 * EE) + h * DD;
  v16bf aq[2];
#pragma unroll
  for (int ks = 0; ks < 2; ks++)
    aq[ks] = make_frag(qrow + ks * 32 + sel * 8, qrow + ks * 32 + 16 + sel * 8);

  v8f o[4];
#pragma unroll
  for (int dt = 0; dt < 4; dt++)
#pragma unroll
    for (int i = 0; i < 8; i++) o[dt][i] = 0.f;
  float rm[8], rl[8];
#pragma unroll
  for (int i = 0; i < 8; i++) { rm[i] = -1e30f; rl[i] = 0.f; }

  const __bf16* kbase = qkv + (size_t)(b * TT) * (3 * EE) + EE + h * DD;
  const __bf16* vtb   = vT + (size_t)(b * HH + h) * DD * TT;

  const int nkt = ((q_base + 15) >> 5) + 1;       // causal: key tiles of 32
  for (int kt = 0; kt < nkt; kt++) {
    const int key0 = kt * 32;

    v8f s[2];
#pragma unroll
    for (int t = 0; t < 2; t++)
#pragma unroll
      for (int i = 0; i < 8; i++) s[t][i] = 0.f;
#pragma unroll
    for (int t = 0; t < 2; t++) {
      const __bf16* krow = kbase + (size_t)(key0 + t * 16 + m16) * (3 * EE);
#pragma unroll
      for (int ks = 0; ks < 2; ks++) {
        const __bf16* kp = krow + ks * 32 + sel * 8;
        v16bf bk = make_frag(kp, kp + 16);
        s[t] = __builtin_amdgcn_wmma_f32_16x16x32_bf16(false, aq[ks], false, bk, (short)0, s[t], false, false);
      }
    }

    float corr[8];
#pragma unroll
    for (int i = 0; i < 8; i++) {
      int q = q_base + sel * 8 + i;
      float v0 = (key0 + m16      <= q) ? s[0][i] * scale : -1e30f;
      float v1 = (key0 + 16 + m16 <= q) ? s[1][i] * scale : -1e30f;
      float mnew = fmaxf(rm[i], rmax16(fmaxf(v0, v1)));
      float p0 = __expf(v0 - mnew), p1 = __expf(v1 - mnew);
      s[0][i] = p0; s[1][i] = p1;
      float psum = rsum16(p0 + p1);
      float c = __expf(rm[i] - mnew);
      rl[i] = rl[i] * c + psum;
      rm[i] = mnew;
      corr[i] = c;
    }
#pragma unroll
    for (int dt = 0; dt < 4; dt++)
#pragma unroll
      for (int i = 0; i < 8; i++) o[dt][i] *= corr[i];

#pragma unroll
    for (int i = 0; i < 8; i++) {
      P[sel * 8 + i][m16]      = f2bf(s[0][i]);
      P[sel * 8 + i][16 + m16] = f2bf(s[1][i]);
    }
    __syncthreads();
    v16bf ap = make_frag(&P[m16][sel * 8], &P[m16][16 + sel * 8]);
    __syncthreads();

#pragma unroll
    for (int dt = 0; dt < 4; dt++) {
      const __bf16* vp = vtb + (size_t)(dt * 16 + m16) * TT + key0 + sel * 8;
      v16bf bv = make_frag(vp, vp + 16);
      o[dt] = __builtin_amdgcn_wmma_f32_16x16x32_bf16(false, ap, false, bv, (short)0, o[dt], false, false);
    }
  }

#pragma unroll
  for (int dt = 0; dt < 4; dt++)
#pragma unroll
    for (int i = 0; i < 8; i++) {
      float val = o[dt][i] / rl[i];
      int row = b * TT + q_base + sel * 8 + i;
      int col = h * DD + dt * 16 + m16;
      y[(size_t)row * EE + col] = f2bf(val);
    }
}

// ---------------- host-side orchestration ---------------------------------------------
extern "C" void kernel_launch(void* const* d_in, const int* in_sizes, int n_in,
                              void* d_out, int out_size, void* d_ws, size_t ws_size,
                              hipStream_t stream) {
  (void)in_sizes; (void)n_in; (void)out_size; (void)ws_size;
  const int*   idx    = (const int*)d_in[0];
  const float* wte    = (const float*)d_in[1];
  const float* wpe    = (const float*)d_in[2];
  const float* g1     = (const float*)d_in[3];
  const float* W_qkv  = (const float*)d_in[4];
  const float* b_qkv  = (const float*)d_in[5];
  const float* W_o    = (const float*)d_in[6];
  const float* b_o    = (const float*)d_in[7];
  const float* g2     = (const float*)d_in[8];
  const float* W_fc   = (const float*)d_in[9];
  const float* b_fc   = (const float*)d_in[10];
  const float* W_down = (const float*)d_in[11];
  const float* b_down = (const float*)d_in[12];
  const float* W_gate = (const float*)d_in[13];
  const float* b_gate = (const float*)d_in[14];
  const float* ln_w   = (const float*)d_in[15];
  const float* ln_b   = (const float*)d_in[16];
  const float* W_head = (const float*)d_in[17];

  char* ws = (char*)d_ws;
  size_t off = 0;
  auto alloc = [&](size_t bytes) -> void* {
    void* p = ws + off;
    off += (bytes + 255) & ~(size_t)255;
    return p;
  };
  // activations
  float*  x    = (float*)alloc((size_t)NTOK * EE * 4);        // residual stream
  __bf16* hbuf = (__bf16*)alloc((size_t)NTOK * EE * 2);       // rmsnorm out
  __bf16* qkvb = (__bf16*)alloc((size_t)NTOK * 3 * EE * 2);   // qkv (bf16)
  __bf16* vTb  = (__bf16*)alloc((size_t)BB * HH * DD * TT * 2);
  __bf16* yb   = (__bf16*)alloc((size_t)NTOK * EE * 2);       // attention out
  __bf16* mfc  = (__bf16*)alloc((size_t)NTOK * 4 * EE * 2);   // gelu(fc) out
  float*  mF   = (float*)alloc((size_t)NTOK * EE * 4);        // down-proj (f32)
  __bf16* mB   = (__bf16*)alloc((size_t)NTOK * EE * 2);       // down-proj (bf16)
  __bf16* ln8  = (__bf16*)alloc((size_t)BB * EE * 2);         // final LN rows
  // bf16 transposed weights (Wt[N][K] per layer)
  __bf16* wtQkv  = (__bf16*)alloc((size_t)LAYERS * EE * 3 * EE * 2);
  __bf16* wtO    = (__bf16*)alloc((size_t)LAYERS * EE * EE * 2);
  __bf16* wtFc   = (__bf16*)alloc((size_t)LAYERS * EE * 4 * EE * 2);
  __bf16* wtDown = (__bf16*)alloc((size_t)LAYERS * 4 * EE * EE * 2);
  __bf16* wtGate = (__bf16*)alloc((size_t)LAYERS * EE * EE * 2);
  __bf16* wtHead = (__bf16*)alloc((size_t)EE * VV * 2);

  // ---- stage 0: weight transpose+convert (one pass; weights then live in L2 as bf16)
  wconv_k<<<dim3(3 * EE / 32, EE / 32, LAYERS), 256, 0, stream>>>(W_qkv,  wtQkv,  EE,     3 * EE);
  wconv_k<<<dim3(EE / 32,     EE / 32, LAYERS), 256, 0, stream>>>(W_o,    wtO,    EE,     EE);
  wconv_k<<<dim3(4 * EE / 32, EE / 32, LAYERS), 256, 0, stream>>>(W_fc,   wtFc,   EE,     4 * EE);
  wconv_k<<<dim3(EE / 32, 4 * EE / 32, LAYERS), 256, 0, stream>>>(W_down, wtDown, 4 * EE, EE);
  wconv_k<<<dim3(EE / 32,     EE / 32, LAYERS), 256, 0, stream>>>(W_gate, wtGate, EE,     EE);
  wconv_k<<<dim3(VV / 32,     EE / 32, 1),      256, 0, stream>>>(W_head, wtHead, EE,     VV);

  const int elemBlocks = (NTOK * EE + 255) / 256;             // 24576
  embed_k<<<elemBlocks, 256, 0, stream>>>(idx, wte, wpe, x);

  for (int l = 0; l < LAYERS; l++) {
    rmsnorm_k<<<NTOK, 256, 0, stream>>>(x, g1 + (size_t)l * EE, hbuf);
    gemm_wmma<EPI_BF16><<<dim3(3 * EE / 64, NTOK / 128), 256, 0, stream>>>(
        hbuf, wtQkv + (size_t)l * EE * 3 * EE, b_qkv + (size_t)l * 3 * EE,
        nullptr, qkvb, nullptr, NTOK, 3 * EE, EE);
    vtrans_k<<<elemBlocks, 256, 0, stream>>>(qkvb, vTb);
    attn_k<<<BB * HH * (TT / 16), 32, 0, stream>>>(qkvb, vTb, yb);
    gemm_wmma<EPI_RESID_F32><<<dim3(EE / 64, NTOK / 128), 256, 0, stream>>>(
        yb, wtO + (size_t)l * EE * EE, b_o + (size_t)l * EE,
        x, nullptr, nullptr, NTOK, EE, EE);
    rmsnorm_k<<<NTOK, 256, 0, stream>>>(x, g2 + (size_t)l * EE, hbuf);
    gemm_wmma<EPI_GELU_BF16><<<dim3(4 * EE / 64, NTOK / 128), 256, 0, stream>>>(
        hbuf, wtFc + (size_t)l * EE * 4 * EE, b_fc + (size_t)l * 4 * EE,
        nullptr, mfc, nullptr, NTOK, 4 * EE, EE);
    gemm_wmma<EPI_F32_BF16><<<dim3(EE / 64, NTOK / 128), 256, 0, stream>>>(
        mfc, wtDown + (size_t)l * 4 * EE * EE, b_down + (size_t)l * EE,
        mF, mB, nullptr, NTOK, EE, 4 * EE);
    gemm_wmma<EPI_GATED><<<dim3(EE / 64, NTOK / 128), 256, 0, stream>>>(
        mB, wtGate + (size_t)l * EE * EE, b_gate + (size_t)l * EE,
        x, nullptr, mF, NTOK, EE, EE);
  }

  lastln_k<<<BB, 256, 0, stream>>>(x, ln_w, ln_b, ln8);
  gemm_wmma<EPI_F32><<<dim3(VV / 64, 1), 256, 0, stream>>>(
      ln8, wtHead, nullptr, (float*)d_out, nullptr, nullptr, BB, VV, EE);
}
